// StatefulCausalAttention_10557029614332
// MI455X (gfx1250) — matrix operand order repaired and checked by reading, not compiled, not run on hardware
//
#include <hip/hip_runtime.h>
#include <hip/hip_bf16.h>

#define SL   128
#define SEQ  2048
#define TT   2304      // total tokens = SEQ + 2*SL
#define DIM  1024
#define NH   16
#define DK   64
#define TP   2312      // score row stride (floats), 2304 + 8 pad
#define TP2  2320      // P row stride (bf16), keeps rows 16B aligned
#define NEGV (-1.0e30f)

typedef __bf16 bf16_t;
typedef __attribute__((ext_vector_type(16))) __bf16 v16bf;
typedef __attribute__((ext_vector_type(8)))  __bf16 v8bf;
typedef __attribute__((ext_vector_type(8)))  float  v8f;

union V16U { v16bf v; v8bf h[2]; };

__device__ __forceinline__ v8bf ld8(const bf16_t* p) {
  return *reinterpret_cast<const v8bf*>(p);
}
__device__ __forceinline__ v16bf mk16(v8bf lo, v8bf hi) {
  V16U u; u.h[0] = lo; u.h[1] = hi; return u.v;
}
// A-matrix (16x32, 16-bit): lane<16 holds K 0..15 (klo=0);
// lane>=16 holds K 16..31 (klo=8).  p points at row base + k-chunk base.
__device__ __forceinline__ v16bf loadA(const bf16_t* p, int klo) {
  return mk16(ld8(p + 2 * klo), ld8(p + 2 * klo + 8));
}
// B-matrix (32x16, 16-bit): lane<16 holds K 0..15, lane>=16 holds K 16..31.
__device__ __forceinline__ v16bf loadB(const bf16_t* p, int khalf) {
  return mk16(ld8(p + khalf), ld8(p + khalf + 8));
}
__device__ __forceinline__ v8f wmma_bf16(v16bf a, v16bf b, v8f c) {
  return __builtin_amdgcn_wmma_f32_16x16x32_bf16(
      /*neg_a=*/false, a, /*neg_b=*/false, b,
      /*c_mod=*/(short)0, c, /*reuse_a=*/false, /*reuse_b=*/false);
}

// ---------------- Stage 0: conversions ----------------
__global__ void cvt_x_kernel(const float* __restrict__ x, bf16_t* __restrict__ xbf, int n) {
  int i = blockIdx.x * blockDim.x + threadIdx.x;
  if (i < n) xbf[i] = (bf16_t)x[i];
}

// src: (H, D, 64) f32 row-major ->  dst: (H, 64, D) bf16 row-major (contraction dim contiguous)
__global__ void cvt_w_kernel(const float* __restrict__ src, bf16_t* __restrict__ dst) {
  int i = blockIdx.x * blockDim.x + threadIdx.x;
  if (i >= NH * DK * DIM) return;
  int h = i >> 16;            // / (64*1024)
  int r = i & 65535;
  int n = r >> 10;            // / 1024
  int d = r & 1023;
  dst[i] = (bf16_t)src[(size_t)h * DIM * DK + (size_t)d * DK + n];
}

// ---------------- Stage 1: projection + l2norm ----------------
// grid: (TT/16, NH, 3 proj), block 128 (4 waves; wave w -> 16 output cols)
__global__ void __launch_bounds__(128)
proj_kernel(const bf16_t* __restrict__ xbf, const bf16_t* __restrict__ wt,
            bf16_t* __restrict__ qb, bf16_t* __restrict__ kb, bf16_t* __restrict__ vt) {
  const int ttile = blockIdx.x;
  const int h     = blockIdx.y;
  const int p     = blockIdx.z;     // 0=q,1=k,2=v
  const int tid   = threadIdx.x;
  const int w     = tid >> 5;
  const int lane  = tid & 31;
  const int rowA  = lane & 15;
  const int hi    = lane >> 4;
  const int klo   = hi * 8;
  const int khalf = hi * 16;
  const int nco   = lane & 15;
  const int mhi   = hi * 8;

  __shared__ float tile[16][68];
  __shared__ float rinv[16];

  const bool is_state = (ttile < (SL / 16)) || (ttile >= ((SL + SEQ) / 16));
  const bf16_t* wbase = wt + ((size_t)((p + (is_state ? 3 : 0)) * NH + h)) * DK * DIM;

  const int t0 = ttile * 16;
  const bf16_t* xrow = xbf + (size_t)(t0 + rowA) * DIM;
  const bf16_t* wrow = wbase + (size_t)(w * 16 + rowA) * DIM;

  v8f acc = {};
  for (int kbs = 0; kbs < DIM; kbs += 32) {
    v16bf A = loadA(xrow + kbs, klo);
    v16bf B = loadB(wrow + kbs, khalf);
    acc = wmma_bf16(A, B, acc);
  }
#pragma unroll
  for (int r = 0; r < 8; r++) tile[mhi + r][w * 16 + nco] = acc[r];
  __syncthreads();

  if (tid < 16) {
    float s = 0.f;
    for (int c = 0; c < DK; c++) { float v = tile[tid][c]; s += v * v; }
    rinv[tid] = 1.0f / fmaxf(sqrtf(s), 1e-12f);
  }
  __syncthreads();

#pragma unroll
  for (int e = 0; e < 8; e++) {
    int idx = tid * 8 + e;
    int row = idx >> 6, col = idx & 63;
    float v = tile[row][col] * rinv[row];
    int t = t0 + row;
    if (p == 0)       qb[((size_t)(h * TT + t)) * DK + col] = (bf16_t)v;
    else if (p == 1)  kb[((size_t)(h * TT + t)) * DK + col] = (bf16_t)v;
    else              vt[((size_t)(h * DK + col)) * TT + t] = (bf16_t)v;  // transposed V
  }
}

// ---------------- Stage 2: attention ----------------
// grid: (TT/16, NH), block 128; dynamic LDS holds full 16xTT score row + bf16 P row.
__global__ void __launch_bounds__(128)
attn_kernel(const bf16_t* __restrict__ qb, const bf16_t* __restrict__ kbuf,
            const bf16_t* __restrict__ vt, const float* __restrict__ scaling,
            float* __restrict__ out) {
  const int qt    = blockIdx.x;
  const int h     = blockIdx.y;
  const int tid   = threadIdx.x;
  const int w     = tid >> 5;
  const int lane  = tid & 31;
  const int rowA  = lane & 15;
  const int hiL   = lane >> 4;
  const int klo   = hiL * 8;
  const int khalf = hiL * 16;
  const int nco   = lane & 15;
  const int mhi   = hiL * 8;

  extern __shared__ char smem[];
  float*  sc   = (float*)smem;                                   // [16][TP]
  bf16_t* pbuf = (bf16_t*)(smem + (size_t)16 * TP * 4);          // [16][TP2]
  float*  red  = (float*)(smem + (size_t)16 * TP * 4 + (size_t)16 * TP2 * 2); // [16][8]
  float*  rmax = red + 128;                                      // [16]
  float*  rsum = rmax + 16;                                      // [16]

  const int   qbase  = qt * 16;
  const int   qlast  = qbase + 15;
  const bool  qstate = (qbase >= SL + SEQ);                      // whole tile (aligned)
  const float scale  = scaling[h];
  const int   kend   = ((qbase + 16) + 31) & ~31;                // causal frontier, mult of 32
  const int   ktlim  = kend >> 4;

  // ---- Phase A: S = Q K^T (scaled, masked) into LDS ----
  const bf16_t* qrow = qb + ((size_t)(h * TT + qbase + rowA)) * DK;
  v16bf A0 = loadA(qrow, klo);
  v16bf A1 = loadA(qrow + 32, klo);

  for (int kt = w; kt < ktlim; kt += 4) {
    const int kbase = kt * 16;
    const bool full_mask = (kbase > qlast) || (qstate && (kbase + 15) < SL);
    if (!full_mask) {
      const bf16_t* krow = kbuf + ((size_t)(h * TT + kbase + rowA)) * DK;
      v8f acc = {};
      acc = wmma_bf16(A0, loadB(krow, khalf), acc);
      acc = wmma_bf16(A1, loadB(krow + 32, khalf), acc);
#pragma unroll
      for (int r = 0; r < 8; r++) {
        int qi  = qbase + mhi + r;
        int key = kbase + nco;
        bool ok = (key <= qi) && !(qstate && (key < SL));
        sc[(size_t)(mhi + r) * TP + key] = ok ? acc[r] * scale : NEGV;
      }
    } else {
#pragma unroll
      for (int r = 0; r < 8; r++)
        sc[(size_t)(mhi + r) * TP + kbase + nco] = NEGV;
    }
  }
  __syncthreads();

  // ---- Phase B: softmax over [0, kend), write bf16 P ----
  {
    const int row = tid >> 3, seg = tid & 7;
    const int seglen = kend >> 3;
    const int s0 = seg * seglen, s1 = s0 + seglen;
    float m = -3.0e38f;
    for (int i = s0; i < s1; i++) m = fmaxf(m, sc[(size_t)row * TP + i]);
    red[row * 8 + seg] = m;
    __syncthreads();
    if (seg == 0) {
      float mm = red[row * 8];
      for (int j = 1; j < 8; j++) mm = fmaxf(mm, red[row * 8 + j]);
      rmax[row] = mm;
    }
    __syncthreads();
    const float mm = rmax[row];
    float sum = 0.f;
    for (int i = s0; i < s1; i++) {
      float pv = __expf(sc[(size_t)row * TP + i] - mm);
      pbuf[(size_t)row * TP2 + i] = (bf16_t)pv;
      sum += pv;
    }
    red[row * 8 + seg] = sum;
    __syncthreads();
    if (seg == 0) {
      float t = 0.f;
      for (int j = 0; j < 8; j++) t += red[row * 8 + j];
      rsum[row] = t;
    }
    __syncthreads();
  }

  // ---- Phase C: O = P V (wave w -> dv cols 16w..16w+15) ----
  {
    const bf16_t* vrow = vt + ((size_t)(h * DK + w * 16 + rowA)) * TT;
    const bf16_t* prow = pbuf + (size_t)rowA * TP2;
    v8f acc = {};
    for (int kbse = 0; kbse < kend; kbse += 32) {
      v16bf A = loadA(prow + kbse, klo);
      v16bf B = loadB(vrow + kbse, khalf);
      acc = wmma_bf16(A, B, acc);
    }
#pragma unroll
    for (int r = 0; r < 8; r++) {
      int rowm = mhi + r;
      out[((size_t)(h * TT + qbase + rowm)) * DK + w * 16 + nco] = acc[r] / rsum[rowm];
    }
  }
}

// ---------------- host launch ----------------
extern "C" void kernel_launch(void* const* d_in, const int* in_sizes, int n_in,
                              void* d_out, int out_size, void* d_ws, size_t ws_size,
                              hipStream_t stream) {
  (void)in_sizes; (void)n_in; (void)out_size; (void)ws_size;
  const float* x       = (const float*)d_in[0];
  const float* scaling = (const float*)d_in[7];

  char* ws = (char*)d_ws;
  // workspace layout (bytes)
  bf16_t* xbf = (bf16_t*)(ws + 0);                       //  4,718,592
  bf16_t* wt  = (bf16_t*)(ws + 4718592);                 // 12,582,912 (6 tensors)
  bf16_t* qb  = (bf16_t*)(ws + 17301504);                //  4,718,592
  bf16_t* kb  = (bf16_t*)(ws + 22020096);                //  4,718,592
  bf16_t* vt  = (bf16_t*)(ws + 26738688);                //  4,718,592 (transposed V)

  const int nx = TT * DIM;
  cvt_x_kernel<<<(nx + 255) / 256, 256, 0, stream>>>(x, xbf, nx);
  const int nw = NH * DK * DIM;
  for (int j = 0; j < 6; j++) {
    cvt_w_kernel<<<(nw + 255) / 256, 256, 0, stream>>>(
        (const float*)d_in[1 + j], wt + (size_t)j * nw);
  }

  dim3 g1(TT / 16, NH, 3);
  proj_kernel<<<g1, 128, 0, stream>>>(xbf, wt, qb, kb, vt);

  const size_t smem = (size_t)16 * TP * 4 + (size_t)16 * TP2 * 2 + 128 * 4 + 16 * 4 + 16 * 4;
  hipFuncSetAttribute(reinterpret_cast<const void*>(attn_kernel),
                      hipFuncAttributeMaxDynamicSharedMemorySize, (int)smem);
  dim3 g2(TT / 16, NH);
  attn_kernel<<<g2, 128, smem, stream>>>(qb, kb, vt, scaling, (float*)d_out);
}
